// JSONGNN_28681791603309
// MI455X (gfx1250) — matrix operand order repaired and checked by reading, not compiled, not run on hardware
//
#include <hip/hip_runtime.h>
#include <hip/hip_bf16.h>

typedef __attribute__((ext_vector_type(16))) _Float16 v16h;
typedef __attribute__((ext_vector_type(8)))  float    v8f;
typedef __attribute__((ext_vector_type(4)))  float    v4f;

// ---------------------------------------------------------------------------
// Degree / normalization kernels
// ---------------------------------------------------------------------------
__global__ void deg_init_kernel(float* __restrict__ deg, int n) {
    int i = blockIdx.x * blockDim.x + threadIdx.x;
    if (i < n) deg[i] = 1.0f;  // self-loop contributes 1 to in-degree
}

__global__ void deg_count_kernel(const int* __restrict__ col, float* __restrict__ deg, int E) {
    int i = blockIdx.x * blockDim.x + threadIdx.x;
    if (i < E) atomicAdd(&deg[col[i]], 1.0f);
}

__global__ void dinv_kernel(float* __restrict__ deg, int n) {
    int i = blockIdx.x * blockDim.x + threadIdx.x;
    if (i < n) {
        float d = deg[i];
        deg[i] = (d > 0.0f) ? rsqrtf(d) : 0.0f;
    }
}

// ---------------------------------------------------------------------------
// WMMA GEMM: C[M, Ntot] = A[M, K] * W[K, Ntot]  (f32 in, f16 multiply, f32 acc)
// Block: 256 threads = 8 waves; each wave computes a 16-row strip.
// Block tile: 128 rows x 64 cols (blockIdx.y selects 64-col slab).
// W is pre-swizzled into LDS in the CDNA5 16-bit B-fragment layout:
//   lane l holds column n = (l&15); halfword h holds k = (h>>3)*16 + (l>>4)*8 + (h&7).
// A fragment (16-bit 16x32 layout): lane l row m=(l&15);
//   halfwords 0..7  -> k = (l>>4)*8 + 0..7
//   halfwords 8..15 -> k = 16 + (l>>4)*8 + 0..7
// ---------------------------------------------------------------------------
template <int K>
__global__ __launch_bounds__(256) void gemm_wmma_kernel(
        const float* __restrict__ A, const float* __restrict__ W,
        float* __restrict__ C, int M, int Ntot) {
    constexpr int KT = K / 32;
    __shared__ __align__(32) _Float16 bfrag[KT][4][32][16];

    const int tid   = threadIdx.x;
    const int nbase = blockIdx.y * 64;

    // Cooperative fill of B fragments (f32 -> f16 convert)
    for (int idx = tid; idx < KT * 2048; idx += 256) {
        int h  = idx & 15;
        int l  = (idx >> 4) & 31;
        int nt = (idx >> 9) & 3;
        int kt = idx >> 11;
        int k  = kt * 32 + ((h >> 3) * 16) + ((l >> 4) * 8) + (h & 7);
        int n  = nbase + nt * 16 + (l & 15);
        bfrag[kt][nt][l][h] = (_Float16)W[(size_t)k * Ntot + n];
    }
    __syncthreads();

    const int lane  = tid & 31;
    const int wave  = tid >> 5;
    const int m0    = blockIdx.x * 128 + wave * 16;
    const int mA    = m0 + (lane & 15);      // row this lane supplies to A fragment
    const int khalf = (lane >> 4) * 8;       // K sub-offset within a 32-wide K tile

    v8f acc[4] = {};

    for (int kt = 0; kt < KT; ++kt) {
        v16h a;
        if (mA < M) {
            const float* ap = A + (size_t)mA * K + kt * 32 + khalf;
            v4f x0 = *(const v4f*)(ap);
            v4f x1 = *(const v4f*)(ap + 4);
            v4f x2 = *(const v4f*)(ap + 16);
            v4f x3 = *(const v4f*)(ap + 20);
#pragma unroll
            for (int j = 0; j < 4; ++j) {
                a[j]      = (_Float16)x0[j];
                a[4 + j]  = (_Float16)x1[j];
                a[8 + j]  = (_Float16)x2[j];
                a[12 + j] = (_Float16)x3[j];
            }
        } else {
#pragma unroll
            for (int j = 0; j < 16; ++j) a[j] = (_Float16)0.0f;
        }
#pragma unroll
        for (int nt = 0; nt < 4; ++nt) {
            v16h b = *(const v16h*)&bfrag[kt][nt][lane][0];
            acc[nt] = __builtin_amdgcn_wmma_f32_16x16x32_f16(
                false, a, false, b, (short)0, acc[nt], false, false);
        }
    }

    // C/D layout: lane l, vgpr j -> m = (l>>4)*8 + j, n = (l&15)
    const int mrow = m0 + (lane >> 4) * 8;
    const int ncol = nbase + (lane & 15);
#pragma unroll
    for (int nt = 0; nt < 4; ++nt) {
#pragma unroll
        for (int j = 0; j < 8; ++j) {
            int m = mrow + j;
            if (m < M) C[(size_t)m * Ntot + ncol + nt * 16] = acc[nt][j];
        }
    }
}

// ---------------------------------------------------------------------------
// Edge scatter: agg[col[e]] += h[row[e]] * dinv[row]*dinv[col]
// One thread per (edge, 4-float chunk); F4 = F/4 chunks per edge.
// ---------------------------------------------------------------------------
template <int F4>
__global__ __launch_bounds__(256) void scatter_kernel(
        const float* __restrict__ h, const int* __restrict__ row,
        const int* __restrict__ col, const float* __restrict__ dinv,
        float* __restrict__ agg, int E) {
    long long idx = (long long)blockIdx.x * blockDim.x + threadIdx.x;
    long long tot = (long long)E * F4;
    if (idx >= tot) return;
    int e = (int)(idx / F4);
    int f = (int)(idx % F4);
    int r = row[e];
    int c = col[e];
    float w = dinv[r] * dinv[c];
    v4f v = *(const v4f*)(h + (size_t)r * (F4 * 4) + f * 4);
    float* dst = agg + (size_t)c * (F4 * 4) + f * 4;
    atomicAdd(dst + 0, w * v[0]);
    atomicAdd(dst + 1, w * v[1]);
    atomicAdd(dst + 2, w * v[2]);
    atomicAdd(dst + 3, w * v[3]);
}

// ---------------------------------------------------------------------------
// Fused self-loop + bias + ReLU (in place on agg)
// ---------------------------------------------------------------------------
template <int F>
__global__ __launch_bounds__(256) void relu_bias_self_kernel(
        float* __restrict__ agg, const float* __restrict__ h,
        const float* __restrict__ dinv, const float* __restrict__ b, int N) {
    long long idx = (long long)blockIdx.x * blockDim.x + threadIdx.x;
    if (idx >= (long long)N * F) return;
    int i = (int)(idx / F);
    int f = (int)(idx % F);
    float sw = dinv[i] * dinv[i];
    float v = agg[idx] + sw * h[idx] + b[f];
    agg[idx] = v > 0.0f ? v : 0.0f;
}

// ---------------------------------------------------------------------------
// Fused self-loop + bias + log_softmax over F=64 classes; one wave32 per row,
// 2 features per lane, wave shuffle reductions.
// ---------------------------------------------------------------------------
__global__ __launch_bounds__(256) void logsoftmax_kernel(
        const float* __restrict__ agg, const float* __restrict__ o,
        const float* __restrict__ dinv, const float* __restrict__ b,
        float* __restrict__ out, int N) {
    int wave = threadIdx.x >> 5;
    int lane = threadIdx.x & 31;
    int nrow = blockIdx.x * (blockDim.x >> 5) + wave;
    if (nrow >= N) return;

    float sw = dinv[nrow] * dinv[nrow];
    size_t base = (size_t)nrow * 64;
    float v0 = agg[base + lane]      + sw * o[base + lane]      + b[lane];
    float v1 = agg[base + lane + 32] + sw * o[base + lane + 32] + b[lane + 32];

    float m = fmaxf(v0, v1);
#pragma unroll
    for (int off = 16; off > 0; off >>= 1)
        m = fmaxf(m, __shfl_xor(m, off, 32));

    float s = __expf(v0 - m) + __expf(v1 - m);
#pragma unroll
    for (int off = 16; off > 0; off >>= 1)
        s += __shfl_xor(s, off, 32);

    float ls = __logf(s);
    out[base + lane]      = v0 - m - ls;
    out[base + lane + 32] = v1 - m - ls;
}

// ---------------------------------------------------------------------------
// Launch
// ---------------------------------------------------------------------------
extern "C" void kernel_launch(void* const* d_in, const int* in_sizes, int n_in,
                              void* d_out, int out_size, void* d_ws, size_t ws_size,
                              hipStream_t stream) {
    (void)n_in; (void)out_size; (void)ws_size;
    const float* x  = (const float*)d_in[0];
    const int*   ei = (const int*)d_in[1];
    const float* W1 = (const float*)d_in[2];
    const float* b1 = (const float*)d_in[3];
    const float* W2 = (const float*)d_in[4];
    const float* b2 = (const float*)d_in[5];
    float* out = (float*)d_out;

    const int N = in_sizes[0] / 256;   // 100000
    const int E = in_sizes[1] / 2;     // 1600000
    const int* row = ei;
    const int* col = ei + E;

    // Workspace layout (256B-aligned slabs)
    char* ws = (char*)d_ws;
    size_t off = 0;
    auto carve = [&](size_t bytes) -> float* {
        float* p = (float*)(ws + off);
        off += (bytes + 255) & ~(size_t)255;
        return p;
    };
    float* dinv = carve((size_t)N * 4);         // deg -> dinv in place
    float* h1   = carve((size_t)N * 128 * 4);   // x @ W1
    float* agg1 = carve((size_t)N * 128 * 4);   // layer-1 aggregation -> h2
    float* o2   = carve((size_t)N * 64 * 4);    // h2 @ W2
    float* agg2 = carve((size_t)N * 64 * 4);    // layer-2 aggregation

    // Zero accumulation buffers every call (deterministic; capture-safe)
    hipMemsetAsync(agg1, 0, (size_t)N * 128 * 4, stream);
    hipMemsetAsync(agg2, 0, (size_t)N * 64 * 4, stream);

    const int T = 256;

    // 1) normalization
    deg_init_kernel<<<(N + T - 1) / T, T, 0, stream>>>(dinv, N);
    deg_count_kernel<<<(E + T - 1) / T, T, 0, stream>>>(col, dinv, E);
    dinv_kernel<<<(N + T - 1) / T, T, 0, stream>>>(dinv, N);

    // 2) h1 = x @ W1   (M=N, K=256, Ntot=128; two 64-col slabs)
    gemm_wmma_kernel<256><<<dim3((N + 127) / 128, 2), T, 0, stream>>>(x, W1, h1, N, 128);

    // 3) edge aggregation, layer 1 (F=128 -> 32 float4 chunks)
    {
        long long tot = (long long)E * 32;
        scatter_kernel<32><<<(unsigned)((tot + T - 1) / T), T, 0, stream>>>(h1, row, col, dinv, agg1, E);
    }

    // 4) self-loop + bias + ReLU  (agg1 becomes h2)
    {
        long long tot = (long long)N * 128;
        relu_bias_self_kernel<128><<<(unsigned)((tot + T - 1) / T), T, 0, stream>>>(agg1, h1, dinv, b1, N);
    }

    // 5) o2 = h2 @ W2   (M=N, K=128, Ntot=64; one 64-col slab)
    gemm_wmma_kernel<128><<<dim3((N + 127) / 128, 1), T, 0, stream>>>(agg1, W2, o2, N, 64);

    // 6) edge aggregation, layer 2 (F=64 -> 16 float4 chunks)
    {
        long long tot = (long long)E * 16;
        scatter_kernel<16><<<(unsigned)((tot + T - 1) / T), T, 0, stream>>>(o2, row, col, dinv, agg2, E);
    }

    // 7) self-loop + bias + log_softmax -> out
    logsoftmax_kernel<<<(N + 7) / 8, T, 0, stream>>>(agg2, o2, dinv, b2, out, N);
}
